// CausalSelfAttention_16750372454541
// MI455X (gfx1250) — matrix operand order repaired and checked by reading, not compiled, not run on hardware
//
#include <hip/hip_runtime.h>
#include <hip/hip_bf16.h>

typedef _Float16 v8h   __attribute__((ext_vector_type(8)));
typedef _Float16 v16h  __attribute__((ext_vector_type(16)));
typedef float    v8f   __attribute__((ext_vector_type(8)));
typedef float    v8ff  __attribute__((ext_vector_type(8)));

constexpr int Bb = 4;
constexpr int Tt = 4096;
constexpr int Cc = 384;
constexpr int Hh = 6;
constexpr int Dd = 64;

__device__ inline v8f wmma16(v16h a, v16h b, v8f c) {
    // v_wmma_f32_16x16x32_f16: D = A(16x32 f16) * B(32x16 f16) + C(16x16 f32)
    return __builtin_amdgcn_wmma_f32_16x16x32_f16(false, a, false, b, (short)0, c, false, false);
}

// A-fragment (16x32 f16). Lane: row m = lane&15, h = lane>>4.
// element e -> k = 16h + e  => one contiguous 16-half run (two 8-half loads).
__device__ inline v16h afrag_f16(const _Float16* row, int h) {
    v8h lo = *(const v8h*)(row + 16 * h);
    v8h hi = *(const v8h*)(row + 16 * h + 8);
    v16h r;
#pragma unroll
    for (int i = 0; i < 8; ++i) { r[i] = lo[i]; r[i + 8] = hi[i]; }
    return r;
}

__device__ inline v16h afrag_f32(const float* row, int h) {
    v8ff lo = *(const v8ff*)(row + 16 * h);
    v8ff hi = *(const v8ff*)(row + 16 * h + 8);
    v16h r;
#pragma unroll
    for (int i = 0; i < 8; ++i) { r[i] = (_Float16)lo[i]; r[i + 8] = (_Float16)hi[i]; }
    return r;
}

// B-fragment (32x16 f16). Lane: col n = lane&15, h = lane>>4.
// element e -> k = 16h + e  => one contiguous 16-half (32B) run per lane.
__device__ inline v16h bfrag_f16(const _Float16* p) {
    return *(const v16h*)p;
}

// ---------------------------------------------------------------------------
// Kernel 0a/0b: one-shot weight conversion f32 -> f16, stored transposed:
// wt[n*K + k] = w[k*N + n]. Q-section columns of w_attn pre-scaled by 1/8.
// ---------------------------------------------------------------------------
__global__ __launch_bounds__(256) void cvt_wattn_kernel(const float* __restrict__ w,
                                                        _Float16* __restrict__ wt) {
    const int idx = (int)(blockIdx.x * blockDim.x + threadIdx.x);
    constexpr int N = 3 * Cc, K = Cc;
    if (idx >= N * K) return;
    const int n = idx / K, k = idx % K;
    const float s = (n < Cc) ? 0.125f : 1.0f;  // fold softmax scale into Q columns
    wt[(size_t)n * K + k] = (_Float16)(w[(size_t)k * N + n] * s);
}

__global__ __launch_bounds__(256) void cvt_wproj_kernel(const float* __restrict__ w,
                                                        _Float16* __restrict__ wt) {
    const int idx = (int)(blockIdx.x * blockDim.x + threadIdx.x);
    constexpr int N = Cc, K = Cc;
    if (idx >= N * K) return;
    const int n = idx / K, k = idx % K;
    wt[(size_t)n * K + k] = (_Float16)(w[(size_t)k * N + n]);
}

// ---------------------------------------------------------------------------
// Kernel 1: qkv = x @ w_attn  (M=B*T=16384, K=384, N=1152)
// Each wave: 16x64 output tile; B-fragments are contiguous v16h loads from the
// transposed f16 weights. Writes Q (pre-scaled), K as [b,h,t,d] f16,
// V transposed as [b,h,d,t] f16.
// ---------------------------------------------------------------------------
__global__ __launch_bounds__(256) void qkv_kernel(const float* __restrict__ x,
                                                  const _Float16* __restrict__ wt,
                                                  _Float16* __restrict__ qb,
                                                  _Float16* __restrict__ kb,
                                                  _Float16* __restrict__ vtb) {
    const int lane = threadIdx.x & 31;
    const int nI = lane & 15;
    const int h = lane >> 4;
    const int wv = (int)((blockIdx.x * blockDim.x + threadIdx.x) >> 5);
    constexpr int NT = (3 * Cc) / 64;   // 18 column tiles
    constexpr int MT = (Bb * Tt) / 16;  // 1024 row tiles
    if (wv >= MT * NT) return;
    const int mt = wv / NT, nt = wv % NT;
    const int mbase = mt * 16, nbase = nt * 64;

    v8f acc[4] = {};
    const float* xrow = x + (size_t)(mbase + nI) * Cc;
    // W^T rows for this wave's 4 column sub-tiles (lane-dependent column)
    const _Float16* wrow[4];
#pragma unroll
    for (int ns = 0; ns < 4; ++ns)
        wrow[ns] = wt + (size_t)(nbase + ns * 16 + nI) * Cc;

    for (int kk = 0; kk < Cc; kk += 32) {
        const v16h a = afrag_f32(xrow + kk, h);
#pragma unroll
        for (int ns = 0; ns < 4; ++ns)
            acc[ns] = wmma16(a, bfrag_f16(wrow[ns] + kk + 16 * h), acc[ns]);
    }

    const int sec = nbase / Cc;               // 0=Q, 1=K, 2=V (64-col tile never straddles)
    const int head = (nbase % Cc) / Dd;
#pragma unroll
    for (int ns = 0; ns < 4; ++ns) {
        const int d = ns * 16 + nI;           // nbase % 64 == 0
#pragma unroll
        for (int v = 0; v < 8; ++v) {
            const int tok = mbase + v + 8 * h;
            const int bI = tok >> 12;          // /T
            const int t = tok & (Tt - 1);
            const float val = acc[ns][v];
            const size_t bh = (size_t)(bI * Hh + head);
            if (sec == 0)      qb[(bh * Tt + t) * Dd + d] = (_Float16)val;
            else if (sec == 1) kb[(bh * Tt + t) * Dd + d] = (_Float16)val;
            else               vtb[(bh * Dd + d) * Tt + t] = (_Float16)val;
        }
    }
}

// ---------------------------------------------------------------------------
// Kernel 2: flash attention. One wave per (b,h,16-row q tile).
// Online softmax over 32-key tiles: 4 WMMAs (QK^T) + 4 WMMAs (P@V) per tile.
// P re-layout (C-layout -> A-fragment) staged through wave-private LDS.
// ---------------------------------------------------------------------------
__global__ __launch_bounds__(256) void attn_kernel(const _Float16* __restrict__ qb,
                                                   const _Float16* __restrict__ kb,
                                                   const _Float16* __restrict__ vtb,
                                                   _Float16* __restrict__ ab) {
    __shared__ __align__(64) _Float16 pst[8][16 * 32];
    const int lane = threadIdx.x & 31;
    const int nI = lane & 15;
    const int h = lane >> 4;
    const int wid = (int)(threadIdx.x >> 5);
    const int wv = (int)blockIdx.x * 8 + wid;
    constexpr int QT = Tt / 16;  // 256
    if (wv >= Bb * Hh * QT) return;
    const int qt = wv % QT;
    const int bh = wv / QT;
    const int qbase = qt * 16;

    const _Float16* Q  = qb  + (size_t)bh * Tt * Dd;
    const _Float16* K  = kb  + (size_t)bh * Tt * Dd;
    const _Float16* Vt = vtb + (size_t)bh * Dd * Tt;
    _Float16* myp = &pst[wid][0];

    // Q A-fragments for d=0..31 and d=32..63 (scale already folded in).
    const _Float16* qrow = Q + (size_t)(qbase + nI) * Dd;
    const v16h qa0 = afrag_f16(qrow, h);
    const v16h qa1 = afrag_f16(qrow + 32, h);

    v8f o[4] = {};
    float mrow[8], lsum[8];
#pragma unroll
    for (int v = 0; v < 8; ++v) { mrow[v] = -1e30f; lsum[v] = 0.f; }

    for (int kt = 0; kt <= qbase + 15; kt += 32) {
        // S (16 q x 32 keys) = Q @ K^T
        const _Float16* krow0 = K + (size_t)(kt + nI) * Dd;
        const _Float16* krow1 = K + (size_t)(kt + 16 + nI) * Dd;
        v8f s0 = {}, s1 = {};
        s0 = wmma16(qa0, bfrag_f16(krow0 + 16 * h), s0);
        s0 = wmma16(qa1, bfrag_f16(krow0 + 32 + 16 * h), s0);
        s1 = wmma16(qa0, bfrag_f16(krow1 + 16 * h), s1);
        s1 = wmma16(qa1, bfrag_f16(krow1 + 32 + 16 * h), s1);

        // causal mask + online softmax (row = v + 8h, cols across 16-lane group)
#pragma unroll
        for (int v = 0; v < 8; ++v) {
            const int qi = qbase + v + 8 * h;
            float a0 = (kt + nI > qi)      ? -1e30f : s0[v];
            float a1 = (kt + 16 + nI > qi) ? -1e30f : s1[v];
            float cm = fmaxf(a0, a1);
            cm = fmaxf(cm, __shfl_xor(cm, 1));
            cm = fmaxf(cm, __shfl_xor(cm, 2));
            cm = fmaxf(cm, __shfl_xor(cm, 4));
            cm = fmaxf(cm, __shfl_xor(cm, 8));
            const float mn = fmaxf(mrow[v], cm);
            const float alpha = __expf(mrow[v] - mn);
            mrow[v] = mn;
            const float p0 = __expf(a0 - mn);
            const float p1 = __expf(a1 - mn);
            lsum[v] = lsum[v] * alpha + p0 + p1;
#pragma unroll
            for (int ds = 0; ds < 4; ++ds) o[ds][v] *= alpha;
            // stage P tile (row-major 16x32) in wave-private LDS
            myp[(v + 8 * h) * 32 + nI]      = (_Float16)p0;
            myp[(v + 8 * h) * 32 + 16 + nI] = (_Float16)p1;
        }
        // same-wave LDS ops are in-order; wait so A-frag reads see all lanes' stores
        asm volatile("s_wait_dscnt 0x0" ::: "memory");
        const _Float16* prow = myp + nI * 32;
        const v16h pa = afrag_f16(prow, h);

        // O += P @ V  (V^T rows are contiguous in t)
#pragma unroll
        for (int ds = 0; ds < 4; ++ds) {
            const _Float16* vrow = Vt + (size_t)(ds * 16 + nI) * Tt + kt;
            o[ds] = wmma16(pa, bfrag_f16(vrow + 16 * h), o[ds]);
        }
    }

    // normalize and write attention output as f16 [b, t, h, d] (== [B,T,C] rows)
    const int bI = bh / Hh, head = bh % Hh;
#pragma unroll
    for (int v = 0; v < 8; ++v) {
        float s = lsum[v];
        s += __shfl_xor(s, 1);
        s += __shfl_xor(s, 2);
        s += __shfl_xor(s, 4);
        s += __shfl_xor(s, 8);
        const float inv = 1.f / s;
        const int t = qbase + v + 8 * h;
        _Float16* orow = ab + ((size_t)(bI * Tt + t) * Hh + head) * Dd;
#pragma unroll
        for (int ds = 0; ds < 4; ++ds)
            orow[ds * 16 + nI] = (_Float16)(o[ds][v] * inv);
    }
}

// ---------------------------------------------------------------------------
// Kernel 3: out = attn_out(f16) @ w_proj^T(f16)  (M=16384, K=384, N=384), f32 out.
// ---------------------------------------------------------------------------
__global__ __launch_bounds__(256) void proj_kernel(const _Float16* __restrict__ ab,
                                                   const _Float16* __restrict__ wt,
                                                   float* __restrict__ out) {
    const int lane = threadIdx.x & 31;
    const int nI = lane & 15;
    const int h = lane >> 4;
    const int wv = (int)((blockIdx.x * blockDim.x + threadIdx.x) >> 5);
    constexpr int NT = Cc / 64;         // 6
    constexpr int MT = (Bb * Tt) / 16;  // 1024
    if (wv >= MT * NT) return;
    const int mt = wv / NT, nt = wv % NT;
    const int mbase = mt * 16, nbase = nt * 64;

    v8f acc[4] = {};
    const _Float16* arow = ab + (size_t)(mbase + nI) * Cc;
    const _Float16* wrow[4];
#pragma unroll
    for (int ns = 0; ns < 4; ++ns)
        wrow[ns] = wt + (size_t)(nbase + ns * 16 + nI) * Cc;

    for (int kk = 0; kk < Cc; kk += 32) {
        const v16h a = afrag_f16(arow + kk, h);
#pragma unroll
        for (int ns = 0; ns < 4; ++ns)
            acc[ns] = wmma16(a, bfrag_f16(wrow[ns] + kk + 16 * h), acc[ns]);
    }
#pragma unroll
    for (int ns = 0; ns < 4; ++ns) {
        const int col = nbase + ns * 16 + nI;
#pragma unroll
        for (int v = 0; v < 8; ++v) {
            const int tok = mbase + v + 8 * h;
            out[(size_t)tok * Cc + col] = acc[ns][v];
        }
    }
}

// ---------------------------------------------------------------------------
extern "C" void kernel_launch(void* const* d_in, const int* in_sizes, int n_in,
                              void* d_out, int out_size, void* d_ws, size_t ws_size,
                              hipStream_t stream) {
    (void)in_sizes; (void)n_in; (void)out_size; (void)ws_size;
    const float* x  = (const float*)d_in[0];   // [B,T,C] f32
    const float* wa = (const float*)d_in[1];   // [C,3C] f32
    const float* wp = (const float*)d_in[2];   // [C,C] f32
    float* out = (float*)d_out;                // [B,T,C] f32

    const size_t plane = (size_t)Bb * Hh * Tt * Dd;  // 6,291,456 halfs (12.6 MB)
    _Float16* qb  = (_Float16*)d_ws;
    _Float16* kb  = qb + plane;
    _Float16* vt  = kb + plane;
    _Float16* ab  = vt + plane;
    _Float16* wta = ab + plane;                      // [1152][384] f16 (w_attn^T)
    _Float16* wtp = wta + (size_t)(3 * Cc) * Cc;     // [384][384]  f16 (w_proj^T)
    // total ~51.5 MB of workspace

    {   // one-shot weight conversion / transpose (f32 -> f16)
        const int na = 3 * Cc * Cc;                  // 442368
        cvt_wattn_kernel<<<(na + 255) / 256, 256, 0, stream>>>(wa, wta);
        const int np = Cc * Cc;                      // 147456
        cvt_wproj_kernel<<<(np + 255) / 256, 256, 0, stream>>>(wp, wtp);
    }
    {   // QKV projection: 1024*18 = 18432 waves, 8 waves/block
        qkv_kernel<<<18432 / 8, 256, 0, stream>>>(x, wta, qb, kb, vt);
    }
    {   // Flash attention: 4*6*256 = 6144 waves
        attn_kernel<<<6144 / 8, 256, 0, stream>>>(qb, kb, vt, ab);
    }
    {   // Output projection: 1024*6 = 6144 waves
        proj_kernel<<<6144 / 8, 256, 0, stream>>>(ab, wtp, out);
    }
}
